// EA_3307124818343
// MI455X (gfx1250) — matrix operand order repaired and checked by reading, not compiled, not run on hardware
//
#include <hip/hip_runtime.h>

// ---------------------------------------------------------------------------
// Types for CDNA5 WMMA
// ---------------------------------------------------------------------------
typedef __attribute__((ext_vector_type(16))) __bf16 v16bf;
typedef __attribute__((ext_vector_type(8)))  __bf16 v8bf;
typedef __attribute__((ext_vector_type(8)))  float  v8f;

#define BATCH 32
#define CH    96
#define HDIM  56
#define HW    3136      // 56*56
#define QDIM  128
#define EDIM  8
#define KTOT  864       // 9 taps * 96 ci
#define NTILE 64        // pixels per block tile (3136 / 64 = 49 tiles)

#if defined(__gfx1250__) && __has_builtin(__builtin_amdgcn_global_load_async_to_lds_b64)
#define EA_USE_ASYNC 1
// exact pointee type required by the builtin: vector of 2 ints (vector_size(8))
typedef int ea_v2i __attribute__((vector_size(8)));
typedef __attribute__((address_space(1))) ea_v2i* ea_gptr;
typedef __attribute__((address_space(3))) ea_v2i* ea_lptr;
#else
#define EA_USE_ASYNC 0
#endif

// float -> bf16, round-to-nearest-even (pure bit manipulation: always compiles)
__device__ __forceinline__ __bf16 f2bf(float f) {
    unsigned u = __builtin_bit_cast(unsigned, f);
    unsigned r = u + 0x7FFFu + ((u >> 16) & 1u);
    unsigned short h = (unsigned short)(r >> 16);
    return __builtin_bit_cast(__bf16, h);
}

__device__ __forceinline__ void ea_wait_async_then_barrier() {
#if EA_USE_ASYNC
#if __has_builtin(__builtin_amdgcn_s_wait_asynccnt)
    __builtin_amdgcn_s_wait_asynccnt(0);
#else
    asm volatile("s_wait_asynccnt 0x0" ::: "memory");
#endif
#endif
    __syncthreads();
}

// ---------------------------------------------------------------------------
// Kernel 1: global-avg-pool -> q @ Wq^T -> @ Wk^T -> softmax  => attn[B,8]
// One block per batch element. Entirely negligible cost.
// ---------------------------------------------------------------------------
__global__ __launch_bounds__(256) void EA_attn_kernel(
        const float* __restrict__ x,
        const float* __restrict__ Wq,
        const float* __restrict__ Wk,
        float* __restrict__ attn) {
    __shared__ float q[CH];
    __shared__ float qd[QDIM];
    const int b = blockIdx.x;
    const int t = threadIdx.x;

    if (t < CH) {
        const float* xp = x + ((size_t)b * CH + t) * HW;
        float s = 0.f;
        for (int i = 0; i < HW; i += 4) {
            float4 v = *(const float4*)(xp + i);
            s += v.x + v.y + v.z + v.w;
        }
        q[t] = s * (1.0f / (float)HW);
    }
    __syncthreads();
    if (t < QDIM) {
        const float* wr = Wq + t * CH;
        float s = 0.f;
        for (int c = 0; c < CH; ++c) s += q[c] * wr[c];
        qd[t] = s;
    }
    __syncthreads();
    if (t == 0) {
        float lg[EDIM];
        float mx = -1e30f;
        for (int e = 0; e < EDIM; ++e) {
            const float* wr = Wk + e * QDIM;
            float s = 0.f;
            for (int d = 0; d < QDIM; ++d) s += qd[d] * wr[d];
            s *= 0.35355339059327373f;   // key_dim^-0.5 = 8^-0.5
            lg[e] = s;
            mx = fmaxf(mx, s);
        }
        float den = 0.f;
        for (int e = 0; e < EDIM; ++e) { lg[e] = __expf(lg[e] - mx); den += lg[e]; }
        float inv = 1.f / den;
        for (int e = 0; e < EDIM; ++e) attn[b * EDIM + e] = lg[e] * inv;
    }
}

// ---------------------------------------------------------------------------
// Kernel 2: x (fp32) -> bf16 copy
// ---------------------------------------------------------------------------
__global__ void EA_cvt_kernel(const float* __restrict__ x,
                              __bf16* __restrict__ xb, int n) {
    int i = blockIdx.x * blockDim.x + threadIdx.x;
    if (i < n) xb[i] = f2bf(x[i]);
}

// ---------------------------------------------------------------------------
// Kernel 3: fold attention into per-batch conv weights (bf16).
//   Weff[b, co, k] with k = tap*96 + ci,  tap = kh*3+kw
//   Weff = sum_e attn[b,e] * Wv[e*96+co, ci, kh, kw]
// ---------------------------------------------------------------------------
__global__ void EA_weff_kernel(const float* __restrict__ Wv,
                               const float* __restrict__ attn,
                               __bf16* __restrict__ Weff) {
    const int total = BATCH * CH * 9 * CH;
    int idx = blockIdx.x * blockDim.x + threadIdx.x;
    if (idx >= total) return;
    int ci  = idx % CH;
    int tmp = idx / CH;
    int tap = tmp % 9;
    tmp /= 9;
    int co = tmp % CH;
    int b  = tmp / CH;
    float s = 0.f;
#pragma unroll
    for (int e = 0; e < EDIM; ++e) {
        s += attn[b * EDIM + e] * Wv[(((size_t)e * CH + co) * CH + ci) * 9 + tap];
    }
    Weff[((size_t)b * CH + co) * KTOT + tap * CH + ci] = f2bf(s);
}

// ---------------------------------------------------------------------------
// Kernel 4: per-batch 3x3 conv as implicit GEMM with bf16 WMMA.
//   GEMM: M = 96 (co), N = 3136 (pixels, 64 per block), K = 864.
//   K staged one spatial tap at a time (full 96-ci slice): 9 stages,
//   each stage = A[96x96]bf16 (18KB) + B[64x96]bf16 (12KB) in LDS,
//   9 WMMAs per wave per stage (3 K-steps x 3 M-tiles).
//   Block: 256 threads = 8 wave32 waves. Wave w -> N-subtile w&3,
//   M-subtiles {w>>2, (w>>2)+2, (w>>2)+4}  => 6x4 = 24 WMMA tiles covered.
//   A tile staged with GLOBAL_LOAD_ASYNC_TO_LDS_B64 (ASYNCcnt) when available.
// ---------------------------------------------------------------------------
__global__ __launch_bounds__(256) void EA_conv_wmma_kernel(
        const __bf16* __restrict__ xb,
        const __bf16* __restrict__ Weff,
        float* __restrict__ out) {
    __shared__ __bf16 As[CH * CH];      // [co][k(ci)] 96x96
    __shared__ __bf16 Bs[NTILE * CH];   // [n][k(ci)]  64x96

    const int b      = blockIdx.y;
    const int p_base = blockIdx.x * NTILE;
    const int t      = threadIdx.x;
    const int lane   = t & 31;
    const int wave   = t >> 5;
    const int n_sub  = wave & 3;     // 0..3
    const int m0     = wave >> 2;    // 0 or 1

    const int col = lane & 15;       // row/col within 16
    const int hi  = lane >> 4;       // lane half selects K sub-block

    v8f acc0 = {};
    v8f acc1 = {};
    v8f acc2 = {};

    const __bf16* wb = Weff + (size_t)b * CH * KTOT;   // this batch's weights

    for (int tap = 0; tap < 9; ++tap) {
        const int kh = tap / 3 - 1;
        const int kw = tap % 3 - 1;

        // prefetch next tap's weight slice while we work on this one
        if (tap < 8 && t < CH)
            __builtin_prefetch(wb + (size_t)t * KTOT + (tap + 1) * CH, 0, 1);

        // ---- stage A: 96 rows x 96 bf16 (192B/row = 24 qwords) ----
#if EA_USE_ASYNC
        for (int i = t; i < CH * 24; i += 256) {
            int row = i / 24;
            int q   = i % 24;
            const __bf16* g = wb + (size_t)row * KTOT + tap * CH + q * 4;
            __bf16*       l = As + row * CH + q * 4;
            __builtin_amdgcn_global_load_async_to_lds_b64(
                (ea_gptr)g, (ea_lptr)l, 0, 0);
        }
#else
        for (int i = t; i < CH * 48; i += 256) {          // dwords
            int row = i / 48;
            int d   = i % 48;
            const unsigned* src =
                (const unsigned*)(wb + (size_t)row * KTOT + tap * CH);
            ((unsigned*)(As + row * CH))[d] = src[d];
        }
#endif
        // ---- stage B: gather shifted window, 96 ci x 64 pixels ----
        for (int i = t; i < CH * NTILE; i += 256) {
            int n = i & (NTILE - 1);   // pixel within tile (coalesced-ish)
            int k = i >> 6;            // ci
            int p = p_base + n;
            int h = p / HDIM, w = p % HDIM;
            int hs = h + kh, ws2 = w + kw;
            __bf16 v = __builtin_bit_cast(__bf16, (unsigned short)0);
            if (hs >= 0 && hs < HDIM && ws2 >= 0 && ws2 < HDIM)
                v = xb[((size_t)b * CH + k) * HW + hs * HDIM + ws2];
            Bs[n * CH + k] = v;
        }
        ea_wait_async_then_barrier();

        // ---- 3 K-steps of 32, 3 M-tiles each ----
        const __bf16* brow  = Bs + (n_sub * 16 + col) * CH;
        const __bf16* arow0 = As + ((m0 + 0) * 16 + col) * CH;
        const __bf16* arow1 = As + ((m0 + 2) * 16 + col) * CH;
        const __bf16* arow2 = As + ((m0 + 4) * 16 + col) * CH;
#pragma unroll
        for (int s = 0; s < 3; ++s) {
            // B fragment: lanes 0-15: K {0..7,16..23}; lanes 16-31: K {8..15,24..31}
            const __bf16* bp = brow + s * 32 + hi * 8;
            v16bf bfrag = __builtin_shufflevector(
                *(const v8bf*)(bp), *(const v8bf*)(bp + 16),
                0,1,2,3,4,5,6,7,8,9,10,11,12,13,14,15);

            // A fragments: lanes 0-15: K {0..7,16..23}; lanes 16-31: K {8..15,24..31}
            const __bf16* ap0 = arow0 + s * 32 + hi * 8;
            const __bf16* ap1 = arow1 + s * 32 + hi * 8;
            const __bf16* ap2 = arow2 + s * 32 + hi * 8;
            v16bf a0 = __builtin_shufflevector(*(const v8bf*)(ap0),
                                               *(const v8bf*)(ap0 + 16),
                0,1,2,3,4,5,6,7,8,9,10,11,12,13,14,15);
            v16bf a1 = __builtin_shufflevector(*(const v8bf*)(ap1),
                                               *(const v8bf*)(ap1 + 16),
                0,1,2,3,4,5,6,7,8,9,10,11,12,13,14,15);
            v16bf a2 = __builtin_shufflevector(*(const v8bf*)(ap2),
                                               *(const v8bf*)(ap2 + 16),
                0,1,2,3,4,5,6,7,8,9,10,11,12,13,14,15);

            acc0 = __builtin_amdgcn_wmma_f32_16x16x32_bf16(
                false, a0, false, bfrag, (short)0, acc0, false, false);
            acc1 = __builtin_amdgcn_wmma_f32_16x16x32_bf16(
                false, a1, false, bfrag, (short)0, acc1, false, false);
            acc2 = __builtin_amdgcn_wmma_f32_16x16x32_bf16(
                false, a2, false, bfrag, (short)0, acc2, false, false);
        }
        __syncthreads();   // protect LDS before next stage overwrites
    }

    // ---- store: C/D layout — VGPR r: M = r (lanes 0-15) / 8+r (lanes 16-31) ----
    const int pcol = p_base + n_sub * 16 + col;
#pragma unroll
    for (int r = 0; r < 8; ++r) {
        int m = r + 8 * hi;
        out[((size_t)b * CH + ((m0 + 0) * 16 + m)) * HW + pcol] = acc0[r];
        out[((size_t)b * CH + ((m0 + 2) * 16 + m)) * HW + pcol] = acc1[r];
        out[((size_t)b * CH + ((m0 + 4) * 16 + m)) * HW + pcol] = acc2[r];
    }
}

// ---------------------------------------------------------------------------
// Launcher
// ---------------------------------------------------------------------------
extern "C" void kernel_launch(void* const* d_in, const int* in_sizes, int n_in,
                              void* d_out, int out_size, void* d_ws, size_t ws_size,
                              hipStream_t stream) {
    const float* x  = (const float*)d_in[0];   // [32,96,56,56]
    const float* Wq = (const float*)d_in[1];   // [128,96]
    const float* Wk = (const float*)d_in[2];   // [8,128]
    const float* Wv = (const float*)d_in[3];   // [768,96,3,3]
    float* out = (float*)d_out;                // [32,96,56,56]

    char* ws = (char*)d_ws;
    // ws layout (all 256B aligned):
    //   [0, 1024)                 : attn  (32*8 fp32)
    //   [1024, 1024+19267584)     : x in bf16 (32*96*3136)
    //   [.., +5308416)            : Weff bf16 (32*96*864)
    float*  attn = (float*)ws;
    __bf16* xbf  = (__bf16*)(ws + 1024);
    __bf16* Weff = (__bf16*)(ws + 1024 + (size_t)BATCH * CH * HW * 2);

    EA_attn_kernel<<<BATCH, 256, 0, stream>>>(x, Wq, Wk, attn);

    int nx = BATCH * CH * HW;
    EA_cvt_kernel<<<(nx + 255) / 256, 256, 0, stream>>>(x, xbf, nx);

    int nw = BATCH * CH * 9 * CH;
    EA_weff_kernel<<<(nw + 255) / 256, 256, 0, stream>>>(Wv, attn, Weff);

    dim3 grid(HW / NTILE, BATCH);   // 49 x 32
    EA_conv_wmma_kernel<<<grid, 256, 0, stream>>>(xbf, Weff, out);
}
